// Attention_25786983645333
// MI455X (gfx1250) — matrix operand order repaired and checked by reading, not compiled, not run on hardware
//
#include <hip/hip_runtime.h>

// ---------------------------------------------------------------------------
// CDNA5 (gfx1250) fused windowed-attention, bf16 WMMA, wave32.
// ---------------------------------------------------------------------------

typedef __attribute__((ext_vector_type(16))) __bf16 v16bf;
typedef __attribute__((ext_vector_type(8)))  __bf16 v8bf;
typedef __attribute__((ext_vector_type(8)))  float  v8f;

#define WIN_N    49      // tokens per window
#define NP       64      // padded tokens
#define DMODEL   256
#define HEADS    8
#define DH       32
#define SCALE    0.17677669529663687f   // 1/sqrt(32)

// WS byte offsets
#define WS_WQKV  0                       // 768*256 bf16 = 393216 B
#define WS_WOUT  393216                  // 256*256 bf16 = 131072 B
#define WS_BIAS  524288                  // 8*64*64 f32  = 131072 B

__device__ __forceinline__ v8f wmma_bf16(v16bf a, v16bf b, v8f c) {
    return __builtin_amdgcn_wmma_f32_16x16x32_bf16(
        /*neg_a=*/false, a, /*neg_b=*/false, b,
        /*c_mod=*/(short)0, c, /*reuse_a=*/false, /*reuse_b=*/false);
}

// A fragment 16x32 bf16: p points at (row base + kbase + kb); two 16B blocks.
__device__ __forceinline__ v16bf load_afrag(const __bf16* p) {
    v8bf lo = *(const v8bf*)p;
    v8bf hi = *(const v8bf*)(p + 8);
    return __builtin_shufflevector(lo, hi,
        0, 1, 2, 3, 4, 5, 6, 7, 8, 9, 10, 11, 12, 13, 14, 15);
}

// ---------------------------------------------------------------------------
// Prep: weights -> bf16, build bias table (h,64,64) with -1e30 padding mask.
// ---------------------------------------------------------------------------
__global__ void prep_kernel(const float* __restrict__ wqkv,
                            const float* __restrict__ wout,
                            const float* __restrict__ rel_emb,
                            const int*   __restrict__ rel_idx,
                            __bf16* __restrict__ wq_bf,
                            __bf16* __restrict__ wo_bf,
                            float*  __restrict__ bias) {
    int idx = blockIdx.x * 256 + threadIdx.x;
    if (idx < 768 * 256) wq_bf[idx] = (__bf16)wqkv[idx];
    if (idx < 256 * 256) wo_bf[idx] = (__bf16)wout[idx];
    if (idx < HEADS * NP * NP) {
        int h = idx >> 12;
        int rem = idx & 4095;
        int i = rem >> 6;
        int j = rem & 63;
        float bv = -1e30f;
        if (i < WIN_N && j < WIN_N)
            bv = rel_emb[rel_idx[i * WIN_N + j] * HEADS + h];
        bias[idx] = bv;
    }
}

// ---------------------------------------------------------------------------
// Fused kernel: one workgroup per window (4096 blocks, 256 threads = 8 waves).
// ---------------------------------------------------------------------------
__global__ __launch_bounds__(256)
void attn_fused_kernel(const float*  __restrict__ x,
                       const __bf16* __restrict__ wq,
                       const __bf16* __restrict__ wo,
                       const float*  __restrict__ bias,
                       float*        __restrict__ out) {
    // LDS layout (144 KB total):
    //   [0      , 32K) Qs  : 64x256 bf16
    //   [32K    , 64K) Ks  : 64x256 bf16
    //   [64K    , 96K) Vt  : [8 heads][32][64] bf16
    //   [96K    ,128K) Xs  : 64x256 bf16   (phase1-2) / Os (phase3-4, overlay)
    //   [128K   ,144K) attn staging: 8 waves x 16x64 bf16
    __shared__ char smem[147456];
    __bf16* Qs = (__bf16*)(smem);
    __bf16* Ks = (__bf16*)(smem + 32768);
    __bf16* Vt = (__bf16*)(smem + 65536);
    __bf16* Xs = (__bf16*)(smem + 98304);
    __bf16* Os = (__bf16*)(smem + 98304);     // overlays Xs after phase 2
    __bf16* stage_base = (__bf16*)(smem + 131072);

    const int tid  = threadIdx.x;
    const int lane = tid & 31;
    const int wave = tid >> 5;
    const int l15  = lane & 15;
    const int hb   = lane >> 4;    // which 16-lane half
    const int kb   = hb * 16;      // A-fragment K sub-block
    const int koff = hb * 16;      // B-fragment K sub-block
    const int win  = blockIdx.x;

    // ---------------- Phase 1: load window -> LDS bf16, pad rows 49..63 -----
    {
        const float4* xr = (const float4*)x;   // 64 float4 per row
        #pragma unroll
        for (int i = 0; i < 16; ++i) {
            int f   = i * 256 + tid;           // 0..4095
            int row = f >> 6;
            int c4  = f & 63;
            float4 v = make_float4(0.f, 0.f, 0.f, 0.f);
            if (row < WIN_N)
                v = xr[((size_t)win * WIN_N + row) * 64 + c4];
            __bf16* dst = Xs + row * DMODEL + c4 * 4;
            dst[0] = (__bf16)v.x; dst[1] = (__bf16)v.y;
            dst[2] = (__bf16)v.z; dst[3] = (__bf16)v.w;
        }
    }
    __syncthreads();

    // ---------------- Phase 2: QKV GEMM (64x256 @ 256x768) ------------------
    // 192 (m,nt) tile jobs over 8 waves; K loop 256 in steps of 32.
    for (int jb = wave; jb < 192; jb += 8) {
        int m  = jb & 3;
        int nt = jb >> 2;                     // 0..47
        v8f acc = {};
        const __bf16* Wn = wq + (nt * 16 + l15) * DMODEL;
        const int arow = m * 16 + l15;
        #pragma unroll
        for (int k0 = 0; k0 < DMODEL; k0 += 32) {
            v16bf a = load_afrag(Xs + arow * DMODEL + k0 + kb);
            v16bf b = *(const v16bf*)(Wn + k0 + koff);
            acc = wmma_bf16(a, b, acc);
        }
        int which = nt >> 4;                  // 0=q 1=k 2=v
        int cin   = (nt & 15) * 16 + l15;     // 0..255 within chunk
        int head  = cin >> 5;
        int c     = cin & 31;
        #pragma unroll
        for (int v = 0; v < 8; ++v) {
            int tok = m * 16 + v + hb * 8;
            __bf16 bv = (__bf16)acc[v];
            if (which == 0)      Qs[tok * DMODEL + cin] = bv;
            else if (which == 1) Ks[tok * DMODEL + cin] = bv;
            else                 Vt[head * 2048 + c * NP + tok] = bv;  // transposed
        }
    }
    __syncthreads();

    // ---------------- Phase 3: attention, one wave per head -----------------
    {
        const int h = wave;
        const float* biasH = bias + h * NP * NP;
        __bf16* stage = stage_base + wave * 1024;   // 16x64 bf16 per wave

        #pragma unroll
        for (int ti = 0; ti < 4; ++ti) {
            // sim row-block: 16 x 64  (dh=32 -> single WMMA K step)
            const int qrow = ti * 16 + l15;
            v16bf aq = load_afrag(Qs + qrow * DMODEL + h * DH + kb);
            float s[4][8];
            #pragma unroll
            for (int tj = 0; tj < 4; ++tj) {
                v16bf bk = *(const v16bf*)(Ks + (tj * 16 + l15) * DMODEL + h * DH + koff);
                v8f accv = {};
                accv = wmma_bf16(aq, bk, accv);
                int jc = tj * 16 + l15;
                #pragma unroll
                for (int v = 0; v < 8; ++v) {
                    int irow = ti * 16 + v + hb * 8;
                    s[tj][v] = accv[v] * SCALE + biasH[irow * NP + jc];
                }
            }
            // softmax per row; row's 16 cols per tile live in one 16-lane half
            #pragma unroll
            for (int v = 0; v < 8; ++v) {
                float mx = fmaxf(fmaxf(s[0][v], s[1][v]), fmaxf(s[2][v], s[3][v]));
                #pragma unroll
                for (int msk = 1; msk < 16; msk <<= 1)
                    mx = fmaxf(mx, __shfl_xor(mx, msk, 32));
                float p0 = __expf(s[0][v] - mx);
                float p1 = __expf(s[1][v] - mx);
                float p2 = __expf(s[2][v] - mx);
                float p3 = __expf(s[3][v] - mx);
                float sm = p0 + p1 + p2 + p3;
                #pragma unroll
                for (int msk = 1; msk < 16; msk <<= 1)
                    sm += __shfl_xor(sm, msk, 32);
                float r = 1.0f / sm;
                int il = v + hb * 8;
                stage[il * NP +  0 + l15] = (__bf16)(p0 * r);
                stage[il * NP + 16 + l15] = (__bf16)(p1 * r);
                stage[il * NP + 32 + l15] = (__bf16)(p2 * r);
                stage[il * NP + 48 + l15] = (__bf16)(p3 * r);
            }
            // attn (16x64) @ V (64x32): K dim = tokens (2 steps), N = dh (2 tiles)
            v8f o0 = {}, o1 = {};
            #pragma unroll
            for (int ks = 0; ks < 2; ++ks) {
                v16bf aa = load_afrag(stage + l15 * NP + ks * 32 + kb);
                v16bf b0 = *(const v16bf*)(Vt + h * 2048 + ( 0 + l15) * NP + ks * 32 + koff);
                v16bf b1 = *(const v16bf*)(Vt + h * 2048 + (16 + l15) * NP + ks * 32 + koff);
                o0 = wmma_bf16(aa, b0, o0);
                o1 = wmma_bf16(aa, b1, o1);
            }
            #pragma unroll
            for (int v = 0; v < 8; ++v) {
                int tok = ti * 16 + v + hb * 8;
                Os[tok * DMODEL + h * DH +      l15] = (__bf16)o0[v];
                Os[tok * DMODEL + h * DH + 16 + l15] = (__bf16)o1[v];
            }
        }
    }
    __syncthreads();

    // ---------------- Phase 4: out projection (64x256 @ 256x256) ------------
    for (int jb = wave; jb < 64; jb += 8) {
        int m  = jb & 3;
        int nt = jb >> 2;                     // 0..15
        v8f acc = {};
        const __bf16* Wn = wo + (nt * 16 + l15) * DMODEL;
        const int arow = m * 16 + l15;
        #pragma unroll
        for (int k0 = 0; k0 < DMODEL; k0 += 32) {
            v16bf a = load_afrag(Os + arow * DMODEL + k0 + kb);
            v16bf b = *(const v16bf*)(Wn + k0 + koff);
            acc = wmma_bf16(a, b, acc);
        }
        #pragma unroll
        for (int v = 0; v < 8; ++v) {
            int tok = m * 16 + v + hb * 8;
            if (tok < WIN_N)
                out[((size_t)win * WIN_N + tok) * DMODEL + nt * 16 + l15] = acc[v];
        }
    }
}

// ---------------------------------------------------------------------------
extern "C" void kernel_launch(void* const* d_in, const int* in_sizes, int n_in,
                              void* d_out, int out_size, void* d_ws, size_t ws_size,
                              hipStream_t stream) {
    const float* x       = (const float*)d_in[0];   // (64,8,8,7,7,256) f32
    const float* wqkv    = (const float*)d_in[1];   // (768,256) f32
    const float* wout    = (const float*)d_in[2];   // (256,256) f32
    const float* rel_emb = (const float*)d_in[3];   // (169,8)   f32
    const int*   rel_idx = (const int*)d_in[4];     // (49,49)   i32

    char* ws = (char*)d_ws;
    __bf16* wq_bf = (__bf16*)(ws + WS_WQKV);
    __bf16* wo_bf = (__bf16*)(ws + WS_WOUT);
    float*  bias  = (float*)(ws + WS_BIAS);

    prep_kernel<<<768, 256, 0, stream>>>(wqkv, wout, rel_emb, rel_idx,
                                         wq_bf, wo_bf, bias);

    attn_fused_kernel<<<4096, 256, 0, stream>>>(x, wq_bf, wo_bf, bias,
                                                (float*)d_out);
}
